// OVHungarianMatcher_6468220747890
// MI455X (gfx1250) — matrix operand order repaired and checked by reading, not compiled, not run on hardware
//
#include <hip/hip_runtime.h>
#include <hip/hip_bf16.h>

// ---------------------------------------------------------------------------
// OV Hungarian matcher cost matrix, fused for MI455X (gfx1250).
//   N = B*Q = 16384 rows, M = 1600 targets, D = 512, C = 91 classes.
//   Dominant work: N x M x D cosine GEMM (26.8 GFLOP) -> v_wmma_f32_16x16x32_f16.
//   LDS-tiled (128x160 block tile, K staged in 64-chunks) to cut L2->WGP fetch
//   traffic from ~1.6 GB to ~377 MB; staging via CDNA5 async global->LDS
//   (global_load_async_to_lds_b128 + s_wait_asynccnt, ASYNCcnt path).
//   Output 105 MB fp32 -> memory-bound tail, non-temporal stores.
// ---------------------------------------------------------------------------

#define N_ROWS 16384
#define M_TGT  1600
#define D_DIM  512
#define C_CLS  91

#define BN  128   // block tile rows (N)
#define BM  160   // block tile cols (M); 1600 = 160 * 10
#define KC  64    // K chunk staged in LDS
#define LDA 72    // padded LDS row stride (halves): 36*r mod 64 hits 16 banks
#define LDB 72

typedef _Float16 v16h __attribute__((ext_vector_type(16)));
typedef _Float16 v8h  __attribute__((ext_vector_type(8)));
typedef float    v8f  __attribute__((ext_vector_type(8)));
typedef int      v4i  __attribute__((ext_vector_type(4)));

union V16 { v16h v; v8h h[2]; };

#if __has_builtin(__builtin_amdgcn_global_load_async_to_lds_b128) && \
    __has_builtin(__builtin_amdgcn_s_wait_asynccnt)
#define USE_ASYNC_LDS 1
#else
#define USE_ASYNC_LDS 0
#endif

typedef __attribute__((address_space(1))) v4i* gv4i_ptr;
typedef __attribute__((address_space(3))) v4i* lv4i_ptr;

__device__ __forceinline__ void stage16(const _Float16* g, _Float16* l) {
#if USE_ASYNC_LDS
  __builtin_amdgcn_global_load_async_to_lds_b128((gv4i_ptr)g, (lv4i_ptr)l, 0, 0);
#else
  *(v8h*)l = *(const v8h*)g;
#endif
}

// ---------------------------------------------------------------------------
// Prep 1: focal class-cost table  clsT[n][c]  (N x 91)
// ---------------------------------------------------------------------------
__global__ __launch_bounds__(256) void prep_class_table(
    const float* __restrict__ logits, float* __restrict__ clsT) {
  size_t idx = (size_t)blockIdx.x * blockDim.x + threadIdx.x;
  if (idx >= (size_t)N_ROWS * C_CLS) return;
  float x = logits[idx];
  float p = 1.0f / (1.0f + __expf(-x));
  float om = 1.0f - p;
  float pos = 0.25f * om * om * (-__logf(p + 1e-8f));
  float neg = 0.75f * p * p * (-__logf(om + 1e-8f));
  clsT[idx] = pos - neg;
}

// ---------------------------------------------------------------------------
// Prep 2: L2-normalize pred_embed rows, convert to f16.  One wave per row.
// ---------------------------------------------------------------------------
__global__ __launch_bounds__(256) void prep_embed_norm(
    const float* __restrict__ pe, _Float16* __restrict__ vnH) {
  int row  = blockIdx.x * 8 + (threadIdx.x >> 5);
  int lane = threadIdx.x & 31;
  const float* src = pe + (size_t)row * D_DIM;
  float vals[16];
  float s = 0.0f;
#pragma unroll
  for (int i = 0; i < 16; ++i) {
    float v = src[lane + i * 32];
    vals[i] = v;
    s += v * v;
  }
#pragma unroll
  for (int off = 16; off > 0; off >>= 1) s += __shfl_xor(s, off, 32);
  float inv = rsqrtf(s);
  _Float16* dst = vnH + (size_t)row * D_DIM;
#pragma unroll
  for (int i = 0; i < 16; ++i) dst[lane + i * 32] = (_Float16)(vals[i] * inv);
}

// ---------------------------------------------------------------------------
// Prep 3: gather text_embeddings[tgt_ids[m]] -> f16 rows (so GEMM B columns
// are contiguous in memory: no transpose loads needed).
// ---------------------------------------------------------------------------
__global__ __launch_bounds__(256) void prep_text_gather(
    const float* __restrict__ te, const int* __restrict__ ids,
    _Float16* __restrict__ teH) {
  int m = blockIdx.x;
  int t = threadIdx.x;
  const float* src = te + (size_t)ids[m] * D_DIM;
  _Float16* dst = teH + (size_t)m * D_DIM;
  dst[t]       = (_Float16)src[t];
  dst[t + 256] = (_Float16)src[t + 256];
}

// ---------------------------------------------------------------------------
// Prep 4: cxcywh -> xyxy + area tables (for pred rows and tgt cols)
// ---------------------------------------------------------------------------
__global__ __launch_bounds__(256) void prep_boxes(
    const float* __restrict__ boxes, float* __restrict__ xy,
    float* __restrict__ area, int count) {
  int i = blockIdx.x * blockDim.x + threadIdx.x;
  if (i >= count) return;
  float4 b = ((const float4*)boxes)[i];
  float x0 = b.x - 0.5f * b.z, y0 = b.y - 0.5f * b.w;
  float x1 = b.x + 0.5f * b.z, y1 = b.y + 0.5f * b.w;
  ((float4*)xy)[i] = make_float4(x0, y0, x1, y1);
  area[i] = (x1 - x0) * (y1 - y0);
}

// ---------------------------------------------------------------------------
// Main: LDS-tiled WMMA GEMM + focal/L1/GIoU epilogue.
//   Block = 256 threads = 8 waves, arranged 4 (N) x 2 (M).
//   Wave tile = 32x80 (2x5 accumulators), block tile = 128 (N) x 160 (M).
//   Grid = (16384/128, 1600/160) = (128, 10).
// ---------------------------------------------------------------------------
__global__ __launch_bounds__(256) void cost_matrix_main(
    const _Float16* __restrict__ vnH,   // [N, 512] f16, normalized
    const _Float16* __restrict__ teH,   // [M, 512] f16, gathered
    const float* __restrict__ clsT,     // [N, 91]
    const int*   __restrict__ tgt_ids,  // [M]
    const float* __restrict__ pred_boxes,  // [N, 4] cxcywh
    const float* __restrict__ tgt_bbox,    // [M, 4] cxcywh
    const float* __restrict__ pXY, const float* __restrict__ pA,
    const float* __restrict__ tXY, const float* __restrict__ tA,
    float* __restrict__ out) {
  __shared__ _Float16 sA[BN * LDA];   // 18.0 KB
  __shared__ _Float16 sB[BM * LDB];   // 22.5 KB

  const int t    = threadIdx.x;
  const int lane = t & 31;
  const int w    = t >> 5;
  const int wrow = w & 3;    // 4 waves along N
  const int wcol = w >> 2;   // 2 waves along M
  const int nblock = blockIdx.x * BN;
  const int mblock = blockIdx.y * BM;
  const int r  = lane & 15;
  const int hi = lane >> 4;

  // Staging: each thread moves one 16B segment per issue; a KC-chunk row is
  // 128 B = 8 segments.  A: 128 rows -> 4 issues/thread; B: 160 rows -> 5.
  const int srow = t >> 3;
  const int sseg = (t & 7) * 8;

  v8f acc[2][5] = {};

  for (int kc = 0; kc < D_DIM; kc += KC) {
    __syncthreads();  // previous chunk fully consumed before overwrite
#pragma unroll
    for (int i = 0; i < 4; ++i) {
      int row = srow + i * 32;
      stage16(vnH + (size_t)(nblock + row) * D_DIM + kc + sseg,
              &sA[row * LDA + sseg]);
    }
#pragma unroll
    for (int i = 0; i < 5; ++i) {
      int row = srow + i * 32;
      stage16(teH + (size_t)(mblock + row) * D_DIM + kc + sseg,
              &sB[row * LDB + sseg]);
    }
#if USE_ASYNC_LDS
    __builtin_amdgcn_s_wait_asynccnt(0);
#endif
    __syncthreads();

#pragma unroll
    for (int ks = 0; ks < KC; ks += 32) {
      V16 Afrag[2], Bfrag[5];
      // A (16x32 f16): lane<16 row r holds K 0..7 & 16..23 of the window,
      // lane>=16 holds K 8..15 & 24..31 -> two b128 LDS reads per subtile.
#pragma unroll
      for (int sn = 0; sn < 2; ++sn) {
        const _Float16* p = &sA[(wrow * 32 + sn * 16 + r) * LDA + hi * 8 + ks];
        Afrag[sn].h[0] = *(const v8h*)(p);
        Afrag[sn].h[1] = *(const v8h*)(p + 16);
      }
      // B (32x16 f16): col r, lane<16 K 0..7 & 16..23, lane>=16 K 8..15 &
      // 24..31 of the staged te row (matches A).
#pragma unroll
      for (int sm = 0; sm < 5; ++sm) {
        const _Float16* p = &sB[(wcol * 80 + sm * 16 + r) * LDB + hi * 8 + ks];
        Bfrag[sm].h[0] = *(const v8h*)(p);
        Bfrag[sm].h[1] = *(const v8h*)(p + 16);
      }
#pragma unroll
      for (int sn = 0; sn < 2; ++sn)
#pragma unroll
        for (int sm = 0; sm < 5; ++sm)
          acc[sn][sm] = __builtin_amdgcn_wmma_f32_16x16x32_f16(
              false, Afrag[sn].v, false, Bfrag[sm].v, (short)0, acc[sn][sm],
              false, false);
    }
  }

  // Epilogue: C/D layout -> lane col = r, VGPR i holds row i + 8*hi.
#pragma unroll
  for (int sm = 0; sm < 5; ++sm) {
    const int m   = mblock + wcol * 80 + sm * 16 + r;
    const int tid = tgt_ids[m];
    const float4 tb = ((const float4*)tgt_bbox)[m];
    const float4 tx = ((const float4*)tXY)[m];
    const float  ta = tA[m];
    const float* clsCol = clsT + tid;  // + n*91 per row
#pragma unroll
    for (int sn = 0; sn < 2; ++sn) {
      v8f c = acc[sn][sm];
#pragma unroll
      for (int i = 0; i < 8; ++i) {
        const int n = nblock + wrow * 32 + sn * 16 + 8 * hi + i;
        const float4 pb = ((const float4*)pred_boxes)[n];
        const float4 px = ((const float4*)pXY)[n];
        const float  pa = pA[n];
        const float cls = clsCol[(size_t)n * C_CLS];
        // L1 on cxcywh
        float l1 = fabsf(pb.x - tb.x) + fabsf(pb.y - tb.y) +
                   fabsf(pb.z - tb.z) + fabsf(pb.w - tb.w);
        // GIoU on xyxy
        float ltx = fmaxf(px.x, tx.x), lty = fmaxf(px.y, tx.y);
        float rbx = fminf(px.z, tx.z), rby = fminf(px.w, tx.w);
        float iw = fmaxf(rbx - ltx, 0.0f), ih = fmaxf(rby - lty, 0.0f);
        float inter = iw * ih;
        float uni = pa + ta - inter;
        float iou = inter / uni;
        float ex0 = fminf(px.x, tx.x), ey0 = fminf(px.y, tx.y);
        float ex1 = fmaxf(px.z, tx.z), ey1 = fmaxf(px.w, tx.w);
        float ew = fmaxf(ex1 - ex0, 0.0f), eh = fmaxf(ey1 - ey0, 0.0f);
        float ae = ew * eh;
        float giou = iou - (ae - uni) / ae;
        // cosine embedding cost (dot already on normalized vectors)
        float emb = 1.0f - c[i];
        float cost = l1 + cls - giou + 0.5f * emb;
        __builtin_nontemporal_store(cost, out + (size_t)n * M_TGT + m);
      }
    }
  }
}

// ---------------------------------------------------------------------------
extern "C" void kernel_launch(void* const* d_in, const int* in_sizes, int n_in,
                              void* d_out, int out_size, void* d_ws, size_t ws_size,
                              hipStream_t stream) {
  const float* pred_logits = (const float*)d_in[0];   // [16,1024,91]
  const float* pred_boxes  = (const float*)d_in[1];   // [16,1024,4]
  const float* pred_embed  = (const float*)d_in[2];   // [16,1024,512]
  const float* text_embed  = (const float*)d_in[3];   // [91,512]
  const int*   tgt_ids     = (const int*)d_in[4];     // [1600]
  const float* tgt_bbox    = (const float*)d_in[5];   // [1600,4]
  float* out = (float*)d_out;                         // [16,1024,1600]

  // Workspace carve-up (256-byte aligned), ~24.8 MB total.
  size_t off = 0;
  auto carve = [&](size_t bytes) {
    void* p = (char*)d_ws + off;
    off += (bytes + 255) & ~(size_t)255;
    return p;
  };
  _Float16* vnH  = (_Float16*)carve((size_t)N_ROWS * D_DIM * 2);
  _Float16* teH  = (_Float16*)carve((size_t)M_TGT * D_DIM * 2);
  float*    clsT = (float*)carve((size_t)N_ROWS * C_CLS * 4);
  float*    pXY  = (float*)carve((size_t)N_ROWS * 4 * 4);
  float*    pA   = (float*)carve((size_t)N_ROWS * 4);
  float*    tXY  = (float*)carve((size_t)M_TGT * 4 * 4);
  float*    tA   = (float*)carve((size_t)M_TGT * 4);
  (void)ws_size; (void)in_sizes; (void)n_in; (void)out_size;

  // Prep passes
  {
    size_t total = (size_t)N_ROWS * C_CLS;
    prep_class_table<<<dim3((unsigned)((total + 255) / 256)), dim3(256), 0, stream>>>(
        pred_logits, clsT);
  }
  prep_embed_norm<<<dim3(N_ROWS / 8), dim3(256), 0, stream>>>(pred_embed, vnH);
  prep_text_gather<<<dim3(M_TGT), dim3(256), 0, stream>>>(text_embed, tgt_ids, teH);
  prep_boxes<<<dim3((N_ROWS + 255) / 256), dim3(256), 0, stream>>>(
      pred_boxes, pXY, pA, N_ROWS);
  prep_boxes<<<dim3((M_TGT + 255) / 256), dim3(256), 0, stream>>>(
      tgt_bbox, tXY, tA, M_TGT);

  // Fused LDS-tiled GEMM + epilogue
  cost_matrix_main<<<dim3(N_ROWS / BN, M_TGT / BM), dim3(256), 0, stream>>>(
      vnH, teH, clsT, tgt_ids, pred_boxes, tgt_bbox, pXY, pA, tXY, tA, out);
}
